// AssembledBlock_2860448219429
// MI455X (gfx1250) — matrix operand order repaired and checked by reading, not compiled, not run on hardware
//
#include <hip/hip_runtime.h>
#include <hip/hip_bf16.h>

// ---------------- problem constants ----------------
#define B_    16
#define C_    32
#define O_    32
#define H_    256
#define W_    256
#define E_    4
#define HID_  8
#define TEMP_ 30.0f
#define PH_   258           // padded NHWC height (1-px zero halo)
#define PW_   258           // padded NHWC width

typedef __attribute__((ext_vector_type(16))) __bf16       v16bf;
typedef __attribute__((ext_vector_type(8)))  float        v8f;
typedef __attribute__((ext_vector_type(4)))  unsigned int v4u;
typedef __attribute__((ext_vector_type(4)))  unsigned int u32x4;
typedef __attribute__((ext_vector_type(4)))  int          i32x4;
typedef __attribute__((ext_vector_type(8)))  int          i32x8;

union Frag16 { v4u u[2]; v16bf v; };   // 32B: two ds_load_b128 -> one WMMA operand

// ---------------- TDM descriptor build + issue (3D tile, LDS row padding) ----------------
// Moves tile_d2 x tile_d1 x tile_d0 (outer..inner) bf16 elements, inner dim contiguous,
// dim0_stride/dim1_stride in elements; pads LDS dest by 16B after every 64B (row pitch 80B).
__device__ __forceinline__ void tdm_load_tile(unsigned lds_addr, const void* gptr,
                                              unsigned tile_d0, unsigned tile_d1, unsigned tile_d2,
                                              unsigned long long stride0, unsigned long long stride1,
                                              unsigned tensor_d0) {
  const unsigned long long ga = (unsigned long long)(uintptr_t)gptr;
  u32x4 g0;
  g0[0] = 1u;                                              // count=1, user descriptor
  g0[1] = lds_addr;                                        // LDS byte address
  g0[2] = (unsigned)(ga & 0xffffffffu);                    // global_addr[31:0]
  g0[3] = (unsigned)((ga >> 32) & 0x01ffffffu) | (2u << 30); // global_addr[56:32] | type=2
  const unsigned tensor_d1 = 1u << 20;                     // never OOB (halo-padded alloc)
  i32x8 g1;
  g1[0] = (int)((1u << 16) | (1u << 20) | (3u << 22) | (3u << 25)); // data_size=2B, pad_en, 16dw interval, 4dw pad
  g1[1] = (int)((tensor_d0 & 0xffffu) << 16);              // [63:48] = tensor_dim0[15:0]
  g1[2] = (int)(((tensor_d0 >> 16) & 0xffffu) | ((tensor_d1 & 0xffffu) << 16));
  g1[3] = (int)(((tensor_d1 >> 16) & 0xffffu) | (tile_d0 << 16));   // tile_dim0
  g1[4] = (int)(tile_d1 | (tile_d2 << 16));                // tile_dim1, tile_dim2
  g1[5] = (int)(stride0 & 0xffffffffu);                    // tensor_dim0_stride[31:0]
  g1[6] = (int)(((stride0 >> 32) & 0xffffu) | ((unsigned)(stride1 & 0xffffu) << 16));
  g1[7] = (int)((stride1 >> 16) & 0xffffffffu);            // tensor_dim1_stride[47:16]
  i32x4 g2; g2[0] = (int)(1u << 20); g2[1] = 0; g2[2] = 0; g2[3] = 0;  // tensor_dim2 large, no dim3
  i32x4 g3; g3[0] = 0; g3[1] = 0; g3[2] = 0; g3[3] = 0;
#if defined(__clang_major__) && (__clang_major__ >= 23)
  i32x8 g4; g4[0]=0; g4[1]=0; g4[2]=0; g4[3]=0; g4[4]=0; g4[5]=0; g4[6]=0; g4[7]=0;
  __builtin_amdgcn_tensor_load_to_lds(g0, g1, g2, g3, g4, 0);
#else
  __builtin_amdgcn_tensor_load_to_lds(g0, g1, g2, g3, 0);
#endif
}

// ---------------- kernel 1: zero the GAP accumulator ----------------
__global__ __launch_bounds__(512)
void zero_gap_kernel(float* __restrict__ gap) { gap[threadIdx.x] = 0.f; }

// ---------------- kernel 2: x fp32 NCHW -> bf16 padded NHWC, fused GAP partial sums ----------------
__global__ __launch_bounds__(256)
void convert_gap_kernel(const float* __restrict__ x, __hip_bfloat16* __restrict__ xb,
                        float* __restrict__ gap) {
  const int b = blockIdx.z, y = blockIdx.y, x0 = blockIdx.x * 64;
  __shared__ float tile[C_][65];                 // 65: odd pitch, conflict-free transpose
  const int tid = threadIdx.x;
  for (int e = tid; e < C_ * 64; e += 256) {     // coalesced NCHW reads
    const int c = e >> 6, xx = e & 63;
    tile[c][xx] = x[(((size_t)b * C_ + c) * H_ + y) * W_ + x0 + xx];
  }
  __syncthreads();
  if (tid < C_) {                                // fused GAP: one partial atomic per (block, c)
    float s = 0.f;
    for (int xx = 0; xx < 64; ++xx) s += tile[tid][xx];
    atomicAdd(&gap[b * C_ + tid], s * (1.0f / (H_ * W_)));
  }
  for (int e = tid; e < C_ * 64; e += 256) {     // coalesced NHWC writes (64B per pixel)
    const int xx = e >> 5, c = e & 31;
    xb[(((size_t)b * PH_ + (y + 1)) * PW_ + (x0 + xx + 1)) * C_ + c] = __float2bfloat16(tile[c][xx]);
  }
}

// ---------------- kernel 3: zero the 1-px halo border of a padded NHWC tensor ----------------
__global__ __launch_bounds__(256)
void border_zero_kernel(__hip_bfloat16* __restrict__ t) {
  const int i = blockIdx.x * 256 + threadIdx.x;  // 16 * 32896 = 526336 exactly
  const int b = i / 32896;
  const int j = i % 32896;
  int y, xx, c;
  if (j < 2 * PW_ * C_) {                        // top/bottom rows
    const int rs = j / (PW_ * C_), r = j % (PW_ * C_);
    xx = r >> 5; c = r & 31; y = rs ? (PH_ - 1) : 0;
  } else {                                       // left/right columns (interior rows)
    const int j2 = j - 2 * PW_ * C_;
    const int cs = j2 / (H_ * C_), r = j2 % (H_ * C_);
    y = 1 + (r >> 5); c = r & 31; xx = cs ? (PW_ - 1) : 0;
  }
  t[(((size_t)b * PH_ + y) * PW_ + xx) * C_ + c] = __float2bfloat16(0.f);
}

// ---------------- kernel 4: control MLP + softmax over experts ----------------
__global__ __launch_bounds__(512)
void coeff_kernel(const float* __restrict__ gap, const float* __restrict__ w_c1,
                  const float* __restrict__ w_c2, float* __restrict__ coeff) {
  const int t = threadIdx.x;                     // 512 = B*O, one (b,o) per thread
  const int b = t >> 5, o = t & 31;
  const float* g = gap + b * C_;
  float h[HID_];
#pragma unroll
  for (int j = 0; j < HID_; ++j) {
    float a = 0.f;
    for (int c = 0; c < C_; ++c) a += w_c1[j * C_ + c] * g[c];
    h[j] = a > 0.f ? a : 0.f;
  }
  float lg[E_], mx = -1e30f;
#pragma unroll
  for (int e = 0; e < E_; ++e) {
    float a = 0.f;
    for (int j = 0; j < HID_; ++j) a += w_c2[(o * E_ + e) * HID_ + j] * h[j];
    lg[e] = a / TEMP_;
    mx = fmaxf(mx, lg[e]);
  }
  float den = 0.f;
#pragma unroll
  for (int e = 0; e < E_; ++e) { lg[e] = __expf(lg[e] - mx); den += lg[e]; }
#pragma unroll
  for (int e = 0; e < E_; ++e) coeff[(b * O_ + o) * E_ + e] = lg[e] / den;
}

// ---------------- kernel 5: expert mixing -> per-sample bf16 weights [set][b][tap][o][c] ----------------
__global__ __launch_bounds__(256)
void assemble_kernel(const float* __restrict__ coeff,
                     const float* __restrict__ w1, const float* __restrict__ w2,
                     const float* __restrict__ w3, __hip_bfloat16* __restrict__ wasm) {
  int i = blockIdx.x * 256 + threadIdx.x;        // 3*B*9*O*C = 442368 exactly
  const int c = i & 31;  i >>= 5;
  const int o = i & 31;  i >>= 5;
  const int tap = i % 9; i /= 9;
  const int b = i & 15;  i >>= 4;
  const int set = i;
  const float* w  = (set == 0) ? w1 : (set == 1) ? w2 : w3;   // [E][O][C][3][3]
  const float* cf = coeff + (b * O_ + o) * E_;
  float a = 0.f;
#pragma unroll
  for (int e = 0; e < E_; ++e)
    a += cf[e] * w[((e * O_ + o) * C_ + c) * 9 + tap];
  wasm[((size_t)set * B_ + b) * (9 * O_ * C_) + (tap * O_ + o) * C_ + c] = __float2bfloat16(a);
}

// ---------------- conv output: bf16 padded NHWC (chained) or fp32 NCHW (final) ----------------
__device__ __forceinline__ void store_px(__hip_bfloat16* out, int b, int ch, int yy, int xx, float v) {
  out[(((size_t)b * PH_ + (yy + 1)) * PW_ + (xx + 1)) * C_ + ch] = __float2bfloat16(v);
}
__device__ __forceinline__ void store_px(float* out, int b, int ch, int yy, int xx, float v) {
  out[(((size_t)b * O_ + ch) * H_ + yy) * W_ + xx] = v;
}

// ---------------- kernel 6: implicit-GEMM 3x3 conv, TDM staging + v_wmma_f32_16x16x32_bf16 ----------------
// M = 16 pixels (along W), N = 16 out-channels (x2 tiles), K = tap*32 + c (9 chunks of 32).
template <typename OutT>
__global__ __launch_bounds__(256)
void dyn_conv_wmma(const __hip_bfloat16* __restrict__ in,    // padded NHWC [B][258][258][32]
                   const __hip_bfloat16* __restrict__ wasm,  // [b][tap][o][c]
                   OutT* __restrict__ out) {
  constexpr int TH = 4, TW = 64;
  constexpr int IY = TH + 2, IX = TW + 2;        // 6 x 66 halo tile
  constexpr int CP = 40;                         // channel pitch (halves): 80B = TDM 64B row + 16B pad
  __shared__ __align__(16) __hip_bfloat16 s_in[IY * IX * CP];  // ~31.7 KB
  __shared__ __align__(16) __hip_bfloat16 s_w[9 * O_ * CP];    // ~23.0 KB

  const int b   = blockIdx.z;
  const int h0  = blockIdx.y * TH;
  const int w0  = blockIdx.x * TW;
  const int tid = threadIdx.x;
  const int wv  = tid >> 5;

  // ---- TDM staging: wave0 pulls the input tile, wave1 the weights ----
  if (wv == 0) {
    const __hip_bfloat16* src = in + (((size_t)b * PH_ + h0) * PW_ + w0) * C_; // halo origin (h0-1+1, w0-1+1)
    tdm_load_tile((unsigned)(uintptr_t)&s_in[0], src,
                  /*d0=*/C_, /*d1=*/IX, /*d2=*/IY,
                  /*stride0=*/C_, /*stride1=*/(unsigned long long)PW_ * C_,
                  /*tensor_d0=*/C_);
    __builtin_amdgcn_s_wait_tensorcnt(0);
  } else if (wv == 1) {
    const __hip_bfloat16* src = wasm + (size_t)b * (9 * O_ * C_);
    tdm_load_tile((unsigned)(uintptr_t)&s_w[0], src,
                  /*d0=*/9 * O_ * C_, /*d1=*/1, /*d2=*/1,
                  /*stride0=*/9 * O_ * C_, /*stride1=*/9 * O_ * C_,
                  /*tensor_d0=*/9 * O_ * C_);
    __builtin_amdgcn_s_wait_tensorcnt(0);
  }
  __syncthreads();

  const int lane = tid & 31;
  const int ml   = lane & 15;                    // A-row (pixel) / B,C column sub-index
  const int hl   = lane >> 4;                    // half-wave: K-split for A/B, M-split for C/D

  for (int it = 0; it < 2; ++it) {
    const int t = wv + 8 * it;                   // 16 M-tiles: 4 rows x 4 segments
    const int r = t >> 2;
    const int s = t & 3;
    v8f acc0 = {0.f, 0.f, 0.f, 0.f, 0.f, 0.f, 0.f, 0.f};
    v8f acc1 = acc0;
#pragma unroll
    for (int tap = 0; tap < 9; ++tap) {
      const int kh = tap / 3;
      const int kw = tap - kh * 3;
      // A: 16 pixels x K=32 channels; lanes 0-15 hold K={0..7,16..23}, lanes 16-31 K={8..15,24..31}
      const __hip_bfloat16* pa = &s_in[((r + kh) * IX + (16 * s + ml + kw)) * CP];
      Frag16 a;
      a.u[0] = *(const v4u*)(pa + 8 * hl);
      a.u[1] = *(const v4u*)(pa + 16 + 8 * hl);
      // B: K=32 channels x 16 out-channels, half-wave K split (same K order as A)
      const __hip_bfloat16* pb0 = &s_w[(tap * O_ + ml) * CP + 8 * hl];
      const __hip_bfloat16* pb1 = &s_w[(tap * O_ + 16 + ml) * CP + 8 * hl];
      Frag16 b0, b1;
      b0.u[0] = *(const v4u*)(pb0);
      b0.u[1] = *(const v4u*)(pb0 + 16);
      b1.u[0] = *(const v4u*)(pb1);
      b1.u[1] = *(const v4u*)(pb1 + 16);
      acc0 = __builtin_amdgcn_wmma_f32_16x16x32_bf16(false, a.v, false, b0.v, (short)0, acc0, false, false);
      acc1 = __builtin_amdgcn_wmma_f32_16x16x32_bf16(false, a.v, false, b1.v, (short)0, acc1, false, false);
    }
    // C/D layout: lanes 0-15 -> N=lane, M=j ; lanes 16-31 -> N=lane-16, M=8+j
    const int yy = h0 + r;
    const int xb = w0 + 16 * s + 8 * hl;
#pragma unroll
    for (int j = 0; j < 8; ++j) {
      store_px(out, b, ml,      yy, xb + j, acc0[j]);
      store_px(out, b, ml + 16, yy, xb + j, acc1[j]);
    }
  }
}

// ---------------- host orchestration ----------------
extern "C" void kernel_launch(void* const* d_in, const int* in_sizes, int n_in,
                              void* d_out, int out_size, void* d_ws, size_t ws_size,
                              hipStream_t stream) {
  (void)in_sizes; (void)n_in; (void)out_size; (void)ws_size;
  const float* x    = (const float*)d_in[0];
  const float* w_c1 = (const float*)d_in[1];
  const float* w_c2 = (const float*)d_in[2];
  const float* w1   = (const float*)d_in[3];
  const float* w2   = (const float*)d_in[4];
  const float* w3   = (const float*)d_in[5];
  float* outp = (float*)d_out;

  // workspace layout (bytes):
  //   [0,4K)        gap    : 512 f32 (atomically accumulated; zeroed each call)
  //   [4K,16K)      coeff  : 2048 f32
  //   [16K,~900K)   wasm   : 3*16*9216 bf16 mixed weights
  //   [1M, +65M)    xb     : bf16 padded NHWC input
  //   [.., +65M)    t1, t2 : bf16 padded NHWC intermediates
  char* ws = (char*)d_ws;
  float* gap            = (float*)(ws);
  float* coeff          = (float*)(ws + 4096);
  __hip_bfloat16* wasm  = (__hip_bfloat16*)(ws + 16384);
  const size_t tsz      = (size_t)B_ * PH_ * PW_ * C_;   // elements per padded NHWC tensor
  __hip_bfloat16* xb    = (__hip_bfloat16*)(ws + (1u << 20));
  __hip_bfloat16* t1    = xb + tsz;
  __hip_bfloat16* t2    = t1 + tsz;

  zero_gap_kernel<<<1, 512, 0, stream>>>(gap);
  convert_gap_kernel<<<dim3(W_ / 64, H_, B_), 256, 0, stream>>>(x, xb, gap);
  {
    const int nblk = (B_ * (2 * PW_ * C_ + 2 * H_ * C_)) / 256;   // 2056
    border_zero_kernel<<<nblk, 256, 0, stream>>>(xb);
    border_zero_kernel<<<nblk, 256, 0, stream>>>(t1);
    border_zero_kernel<<<nblk, 256, 0, stream>>>(t2);
  }
  coeff_kernel<<<1, 512, 0, stream>>>(gap, w_c1, w_c2, coeff);
  assemble_kernel<<<(3 * B_ * 9 * O_ * C_) / 256, 256, 0, stream>>>(coeff, w1, w2, w3, wasm);

  dim3 grid(W_ / 64, H_ / 4, B_);
  const size_t wstep = (size_t)B_ * 9 * O_ * C_;
  dyn_conv_wmma<__hip_bfloat16><<<grid, 256, 0, stream>>>(xb, wasm,             t1);
  dyn_conv_wmma<__hip_bfloat16><<<grid, 256, 0, stream>>>(t1, wasm + wstep,     t2);
  dyn_conv_wmma<float         ><<<grid, 256, 0, stream>>>(t2, wasm + 2 * wstep, outp);
}
